// Mesh2Mesh_26250840113769
// MI455X (gfx1250) — matrix operand order repaired and checked by reading, not compiled, not run on hardware
//
#include <hip/hip_runtime.h>

typedef __attribute__((ext_vector_type(16))) _Float16 v16h;
typedef __attribute__((ext_vector_type(8)))  _Float16 v8h;
typedef __attribute__((ext_vector_type(8)))  float    v8f;
typedef __attribute__((ext_vector_type(4)))  float    v4f;

#define D_   128
#define SK1  392   // 384 + 8 f16 pad: avoids LDS bank conflicts on B-fragment loads
#define SK2  264   // 256 + 8
#define KNBR 8

__device__ __forceinline__ v16h pack16(const float* c0, const float* c1) {
  v16h a;
#pragma unroll
  for (int i = 0; i < 8; i++) { a[i] = (_Float16)c0[i]; a[8 + i] = (_Float16)c1[i]; }
  return a;
}

__device__ __forceinline__ v16h lds_b16(const _Float16* p) {
  v8h lo = *(const v8h*)p;
  v8h hi = *(const v8h*)(p + 16);
  v16h b;
#pragma unroll
  for (int i = 0; i < 8; i++) { b[i] = lo[i]; b[8 + i] = hi[i]; }
  return b;
}

// ---------------------------------------------------------------------------
// Edge update: delta_e = LN(tanh(cat[bond,node[src],node[dst]] @ W1))
// outB = bond + delta_e ; ws = delta_e (if provided)
// One wave = 16 edges x 128 cols; K=384 in 12 chunks of 32 (f16 WMMA, f32 acc)
// ---------------------------------------------------------------------------
__global__ void __launch_bounds__(256, 2) edge_update_kernel(
    const float* __restrict__ bond, const float* __restrict__ node,
    const float* __restrict__ W1,   const float* __restrict__ g1,
    const float* __restrict__ b1,   const int*   __restrict__ epair,
    float* __restrict__ outB, float* __restrict__ ws, int E, int hasWs)
{
  extern __shared__ _Float16 smem[];
  _Float16* w1t = smem;  // [128][SK1], w1t[n][k] = W1[k][n] in f16
  for (int idx = threadIdx.x; idx < 128 * 384; idx += 256) {
    int n = idx / 384, k = idx - n * 384;
    w1t[n * SK1 + k] = (_Float16)W1[(size_t)k * 128 + n];
  }
  __syncthreads();

  const int wave  = threadIdx.x >> 5;
  const int lane  = threadIdx.x & 31;
  const int laneN = lane & 15;
  const int half  = lane >> 4;
  const long tile = (long)blockIdx.x * 8 + wave;
  const long e0   = tile * 16;
  if (e0 >= E) return;

  long arow = e0 + laneN; if (arow >= E) arow = E - 1;
  const int   src  = epair[arow * 2 + 0];
  const int   dst  = epair[arow * 2 + 1];
  const float* bptr = bond + arow * (long)D_;
  const float* sptr = node + (long)src * D_;
  const float* dptr = node + (long)dst * D_;

  v8f acc[8] = {};
#pragma unroll
  for (int ck = 0; ck < 12; ck++) {
    const int base = ck * 32 + (half ? 8 : 0);
    float c0[8], c1[8];
    {
      int f = base;
      const float* p = (f < 128) ? (bptr + f) : ((f < 256) ? (sptr + f - 128) : (dptr + f - 256));
      v4f x = *(const v4f*)p, y = *(const v4f*)(p + 4);
#pragma unroll
      for (int i = 0; i < 4; i++) { c0[i] = x[i]; c0[4 + i] = y[i]; }
    }
    {
      int f = base + 16;
      const float* p = (f < 128) ? (bptr + f) : ((f < 256) ? (sptr + f - 128) : (dptr + f - 256));
      v4f x = *(const v4f*)p, y = *(const v4f*)(p + 4);
#pragma unroll
      for (int i = 0; i < 4; i++) { c1[i] = x[i]; c1[4 + i] = y[i]; }
    }
    v16h a = pack16(c0, c1);
    const int kb = ck * 32 + (half ? 8 : 0);
#pragma unroll
    for (int t = 0; t < 8; t++) {
      v16h b = lds_b16(&w1t[(t * 16 + laneN) * SK1 + kb]);
      acc[t] = __builtin_amdgcn_wmma_f32_16x16x32_f16(false, a, false, b,
                                                      (short)0, acc[t], false, false);
    }
  }

  // tanh + in-register LayerNorm (row m = vgpr m%8 of one lane-half)
  float sum[8], sq[8];
#pragma unroll
  for (int v = 0; v < 8; v++) { sum[v] = 0.f; sq[v] = 0.f; }
#pragma unroll
  for (int t = 0; t < 8; t++)
#pragma unroll
    for (int v = 0; v < 8; v++) {
      float th = tanhf(acc[t][v]);
      acc[t][v] = th; sum[v] += th; sq[v] += th * th;
    }
#pragma unroll
  for (int m = 1; m < 16; m <<= 1) {
#pragma unroll
    for (int v = 0; v < 8; v++) {
      sum[v] += __shfl_xor(sum[v], m, 32);
      sq[v]  += __shfl_xor(sq[v],  m, 32);
    }
  }
  float mean[8], rstd[8];
#pragma unroll
  for (int v = 0; v < 8; v++) {
    float mu  = sum[v] * (1.f / 128.f);
    float var = sq[v] * (1.f / 128.f) - mu * mu;
    mean[v] = mu; rstd[v] = rsqrtf(var + 1e-5f);
  }
#pragma unroll
  for (int t = 0; t < 8; t++) {
    const int col = t * 16 + laneN;
    const float gc = g1[col], bc = b1[col];
#pragma unroll
    for (int v = 0; v < 8; v++) {
      long row = e0 + v + 8 * half;
      if (row < E) {
        float dlt = (acc[t][v] - mean[v]) * rstd[v] * gc + bc;
        size_t o = (size_t)row * D_ + col;
        outB[o] = bond[o] + dlt;
        if (hasWs) ws[o] = dlt;
      }
    }
  }
}

// ---------------------------------------------------------------------------
// Node update: agg = mean over K nbrs of delta_e; delta_n = LN(tanh(cat@W2))
// USE_WS: delta read from ws; else reconstructed as outB - bond.
// ---------------------------------------------------------------------------
template <bool USE_WS>
__global__ void __launch_bounds__(256, 2) node_update_kernel(
    const float* __restrict__ node, const float* __restrict__ W2,
    const float* __restrict__ g2,   const float* __restrict__ b2,
    const float* __restrict__ numl, const int*   __restrict__ nbrTbl,
    const float* __restrict__ ws,   const float* __restrict__ outB,
    const float* __restrict__ bond, float* __restrict__ outN, int N, int E)
{
  extern __shared__ _Float16 smem[];
  _Float16* w2t = smem;  // [128][SK2]
  for (int idx = threadIdx.x; idx < 128 * 256; idx += 256) {
    int n = idx >> 8, k = idx & 255;
    w2t[n * SK2 + k] = (_Float16)W2[(size_t)k * 128 + n];
  }
  __syncthreads();

  const int wave  = threadIdx.x >> 5;
  const int lane  = threadIdx.x & 31;
  const int laneN = lane & 15;
  const int half  = lane >> 4;
  const long tile = (long)blockIdx.x * 8 + wave;
  const long n0   = tile * 16;
  if (n0 >= N) return;

  long arow = n0 + laneN; if (arow >= N) arow = N - 1;
  int nbrs[KNBR];
#pragma unroll
  for (int j = 0; j < KNBR; j++) nbrs[j] = nbrTbl[arow * KNBR + j];
  const float inv  = 1.f / numl[arow];
  const float* nptr = node + arow * (long)D_;

  auto loadchunk = [&](float* o, int f) {
    if (f < 128) {
      v4f x = *(const v4f*)(nptr + f), y = *(const v4f*)(nptr + f + 4);
#pragma unroll
      for (int i = 0; i < 4; i++) { o[i] = x[i]; o[4 + i] = y[i]; }
    } else {
      const int g = f - 128;
      float s[8];
#pragma unroll
      for (int i = 0; i < 8; i++) s[i] = 0.f;
#pragma unroll
      for (int j = 0; j < KNBR; j++) {
        long id = nbrs[j];
        if (id < E) {                     // id == E is the padded zero row
          if (USE_WS) {
            const float* p = ws + id * (long)D_ + g;
            v4f x = *(const v4f*)p, y = *(const v4f*)(p + 4);
#pragma unroll
            for (int i = 0; i < 4; i++) { s[i] += x[i]; s[4 + i] += y[i]; }
          } else {
            const float* p = outB + id * (long)D_ + g;
            const float* q = bond + id * (long)D_ + g;
            v4f x = *(const v4f*)p, y = *(const v4f*)(p + 4);
            v4f xb = *(const v4f*)q, yb = *(const v4f*)(q + 4);
#pragma unroll
            for (int i = 0; i < 4; i++) { s[i] += x[i] - xb[i]; s[4 + i] += y[i] - yb[i]; }
          }
        }
      }
#pragma unroll
      for (int i = 0; i < 8; i++) o[i] = s[i] * inv;
    }
  };

  v8f acc[8] = {};
#pragma unroll
  for (int ck = 0; ck < 8; ck++) {
    const int base = ck * 32 + (half ? 8 : 0);
    float c0[8], c1[8];
    loadchunk(c0, base);
    loadchunk(c1, base + 16);
    v16h a = pack16(c0, c1);
    const int kb = ck * 32 + (half ? 8 : 0);
#pragma unroll
    for (int t = 0; t < 8; t++) {
      v16h b = lds_b16(&w2t[(t * 16 + laneN) * SK2 + kb]);
      acc[t] = __builtin_amdgcn_wmma_f32_16x16x32_f16(false, a, false, b,
                                                      (short)0, acc[t], false, false);
    }
  }

  float sum[8], sq[8];
#pragma unroll
  for (int v = 0; v < 8; v++) { sum[v] = 0.f; sq[v] = 0.f; }
#pragma unroll
  for (int t = 0; t < 8; t++)
#pragma unroll
    for (int v = 0; v < 8; v++) {
      float th = tanhf(acc[t][v]);
      acc[t][v] = th; sum[v] += th; sq[v] += th * th;
    }
#pragma unroll
  for (int m = 1; m < 16; m <<= 1) {
#pragma unroll
    for (int v = 0; v < 8; v++) {
      sum[v] += __shfl_xor(sum[v], m, 32);
      sq[v]  += __shfl_xor(sq[v],  m, 32);
    }
  }
  float mean[8], rstd[8];
#pragma unroll
  for (int v = 0; v < 8; v++) {
    float mu  = sum[v] * (1.f / 128.f);
    float var = sq[v] * (1.f / 128.f) - mu * mu;
    mean[v] = mu; rstd[v] = rsqrtf(var + 1e-5f);
  }
#pragma unroll
  for (int t = 0; t < 8; t++) {
    const int col = t * 16 + laneN;
    const float gc = g2[col], bc = b2[col];
#pragma unroll
    for (int v = 0; v < 8; v++) {
      long row = n0 + v + 8 * half;
      if (row < N) {
        float dlt = (acc[t][v] - mean[v]) * rstd[v] * gc + bc;
        size_t o = (size_t)row * D_ + col;
        outN[o] = node[o] + dlt;
      }
    }
  }
}

extern "C" void kernel_launch(void* const* d_in, const int* in_sizes, int n_in,
                              void* d_out, int out_size, void* d_ws, size_t ws_size,
                              hipStream_t stream) {
  const float* bond  = (const float*)d_in[0];
  const float* node  = (const float*)d_in[1];
  const float* W1    = (const float*)d_in[2];
  const float* g1    = (const float*)d_in[3];
  const float* b1    = (const float*)d_in[4];
  const float* W2    = (const float*)d_in[5];
  const float* g2    = (const float*)d_in[6];
  const float* b2    = (const float*)d_in[7];
  const float* numl  = (const float*)d_in[8];
  const int*   epair = (const int*)d_in[9];
  const int*   nbrT  = (const int*)d_in[10];

  const int E = in_sizes[0] / D_;
  const int N = in_sizes[1] / D_;

  float* outB = (float*)d_out;
  float* outN = outB + (size_t)E * D_;

  const bool useWs = ws_size >= (size_t)E * D_ * sizeof(float);
  float* ws = useWs ? (float*)d_ws : nullptr;

  const int tilesE = (E + 15) / 16;
  const int tilesN = (N + 15) / 16;
  dim3 blk(256);
  dim3 gE((tilesE + 7) / 8);
  dim3 gN((tilesN + 7) / 8);
  size_t ldsA = (size_t)128 * SK1 * 2;  // 98 KB f16 W1^T
  size_t ldsB = (size_t)128 * SK2 * 2;  // 66 KB f16 W2^T

  edge_update_kernel<<<gE, blk, ldsA, stream>>>(bond, node, W1, g1, b1, epair,
                                                outB, ws, E, useWs ? 1 : 0);
  if (useWs)
    node_update_kernel<true><<<gN, blk, ldsB, stream>>>(node, W2, g2, b2, numl, nbrT,
                                                        ws, outB, bond, outN, N, E);
  else
    node_update_kernel<false><<<gN, blk, ldsB, stream>>>(node, W2, g2, b2, numl, nbrT,
                                                         ws, outB, bond, outN, N, E);
}
